// DownstreamModel_44985487458623
// MI455X (gfx1250) — compile-verified, matrix operand order checked by reading, not run on hardware
//
#include <hip/hip_runtime.h>
#include <hip/hip_bf16.h>
#include <math.h>

// ---------------------------------------------------------------------------
// Problem constants (match reference)
// ---------------------------------------------------------------------------
#define N_NODES  50000
#define N_EDGES  1000000
#define N_GRAPHS 64
#define F_IN     113
#define K1P      128     // F_IN padded to WMMA K multiple (zero tail)
#define HDIM     512
#define Z_DIM    518

// ---------------------------------------------------------------------------
// WMMA types (CDNA5 gfx1250, wave32)
// ---------------------------------------------------------------------------
typedef __attribute__((ext_vector_type(16))) __bf16         v16bf;
typedef __attribute__((ext_vector_type(16))) unsigned short v16us;
typedef __attribute__((ext_vector_type(8)))  float          v8f;

union BF16Frag {
  unsigned       w[8];
  v16us          u;
  v16bf          b;
};

__device__ __forceinline__ unsigned short f32_to_bf16_bits(float f) {
  unsigned u = __builtin_bit_cast(unsigned, f);
  unsigned r = u + 0x7fffu + ((u >> 16) & 1u);   // round-to-nearest-even
  return (unsigned short)(r >> 16);
}

// Pack hi-16 of two f32 into one dword of 2 bf16 (truncation, 1 v_perm).
__device__ __forceinline__ unsigned pack_bf16_trunc(float lo, float hi) {
#if __has_builtin(__builtin_amdgcn_perm)
  return __builtin_amdgcn_perm(__builtin_bit_cast(unsigned, hi),
                               __builtin_bit_cast(unsigned, lo), 0x07060302u);
#else
  return (__builtin_bit_cast(unsigned, lo) >> 16) |
         (__builtin_bit_cast(unsigned, hi) & 0xffff0000u);
#endif
}

// ---------------------------------------------------------------------------
// Fragment loaders. NO bounds checks: K (=Kp) must be a multiple of 32 and
// buffers zero-padded. Per ISA 7.12.2, a lane's 16 elements are two runs of
// 8 consecutive K values: [k0+half*8, +8) and [k0+16+half*8, +8).
// A (16x32, M x K): lanes 0-15 -> M=lane; lanes 16-31 -> M=lane-16, K+8.
// B (32x16, K x N): identical addressing with N in place of M (W row-major
// [512 x Kp] so B[k][n] = W[n][k] is contiguous in k). Same loader serves a
// bf16 A operand.
// ---------------------------------------------------------------------------
__device__ __forceinline__ v16bf load_frag_f32cvt(const float* __restrict__ A,
                                                  int row, int Kp, int k0, int half) {
  const float* rp = A + (size_t)row * Kp + k0 + half * 8;
  const float4 p0 = *(const float4*)(rp + 0);
  const float4 p1 = *(const float4*)(rp + 4);
  const float4 q0 = *(const float4*)(rp + 16);
  const float4 q1 = *(const float4*)(rp + 20);
  BF16Frag f;
  f.w[0] = pack_bf16_trunc(p0.x, p0.y);
  f.w[1] = pack_bf16_trunc(p0.z, p0.w);
  f.w[2] = pack_bf16_trunc(p1.x, p1.y);
  f.w[3] = pack_bf16_trunc(p1.z, p1.w);
  f.w[4] = pack_bf16_trunc(q0.x, q0.y);
  f.w[5] = pack_bf16_trunc(q0.z, q0.w);
  f.w[6] = pack_bf16_trunc(q1.x, q1.y);
  f.w[7] = pack_bf16_trunc(q1.z, q1.w);
  return f.b;
}

__device__ __forceinline__ v16bf load_frag_bf16(const unsigned short* __restrict__ W,
                                                int row, int Kp, int k0, int half) {
  const unsigned short* rp = W + (size_t)row * Kp + k0 + half * 8;
  const uint4 r0 = *(const uint4*)(rp + 0);   // k0+half*8 .. +7
  const uint4 r1 = *(const uint4*)(rp + 16);  // +16 .. +23
  BF16Frag f;
  f.w[0] = r0.x; f.w[1] = r0.y; f.w[2] = r0.z; f.w[3] = r0.w;
  f.w[4] = r1.x; f.w[5] = r1.y; f.w[6] = r1.z; f.w[7] = r1.w;
  return f.b;
}

// ---------------------------------------------------------------------------
// Utility kernels
// ---------------------------------------------------------------------------
__global__ void zero_f32(float* __restrict__ p, int n) {
  int i = blockIdx.x * blockDim.x + threadIdx.x;
  if (i < n) p[i] = 0.0f;
}

// f32 [rows x Kin] -> zero-padded bf16 [rows x Kp]
__global__ void cvt_pad_bf16(const float* __restrict__ w, unsigned short* __restrict__ o,
                             int rows, int Kin, int Kp) {
  int i = blockIdx.x * blockDim.x + threadIdx.x;
  if (i >= rows * Kp) return;
  int r = i / Kp, k = i - r * Kp;
  o[i] = (k < Kin) ? f32_to_bf16_bits(w[(size_t)r * Kin + k]) : (unsigned short)0;
}

// ---------------------------------------------------------------------------
// Edge scatter: layer 1 (F_IN feats into stride-K1P buffer) + degree count
// ---------------------------------------------------------------------------
__global__ __launch_bounds__(128) void scatter_edges_1(
    const float* __restrict__ x, const int* __restrict__ src,
    const int* __restrict__ dst, float* __restrict__ msg,
    float* __restrict__ cnt) {
  int e = blockIdx.x;
  int s = src[e], d = dst[e];
  int t = threadIdx.x;
  if (t == 0) atomicAdd(&cnt[d], 1.0f);
  if (t < F_IN) atomicAdd(&msg[(size_t)d * K1P + t], x[(size_t)s * F_IN + t]);
}

// Edge scatter: layer 2. Gathers bf16 h (halves the dominant 2 GB read
// stream), exact-converts to f32, atomic-adds into f32 msg buffer.
__global__ __launch_bounds__(256) void scatter_edges_2(
    const unsigned short* __restrict__ hb, const int* __restrict__ src,
    const int* __restrict__ dst, float* __restrict__ msg) {
  int e = blockIdx.x;
  int s = src[e], d = dst[e];
  for (int f = threadIdx.x; f < HDIM; f += 256) {
    float v = __builtin_bit_cast(float, (unsigned)hb[(size_t)s * HDIM + f] << 16);
    atomicAdd(&msg[(size_t)d * HDIM + f], v);
  }
}

// ---------------------------------------------------------------------------
// Fused SAGE layer:  out = relu(l2norm( (msg/cnt)@Wl^T + bias + xin@Wr^T ))
// Block = 256 threads (8 wave32); block owns rows [16*blockIdx.x, +16) x all
// 512 cols; wave w owns column tiles [64w, 64w+64). The mean divide is hoisted
// out of the GEMM (per-row scale of the accumulators between the two paths).
// Kp must be a multiple of 32 with zero-padded operands (no bounds checks).
// ---------------------------------------------------------------------------
__global__ __launch_bounds__(256) void sage_layer_wmma(
    const float* __restrict__ msg,            // [N, Kp] f32 neighbor sums
    const unsigned short* __restrict__ xinb,  // [N, Kp] bf16 self features
    const unsigned short* __restrict__ Wl,    // [512, Kp] bf16
    const unsigned short* __restrict__ Wr,    // [512, Kp] bf16
    const float* __restrict__ bias,           // [512]
    const float* __restrict__ cnt,            // [N]
    float* __restrict__ outf,                 // [N, 512] f32 (optional)
    unsigned short* __restrict__ outb,        // [N, 512] bf16 (optional)
    int Kp) {
  __shared__ float lds_inv[16];
  __shared__ float lds_nrm[16];

  const int m0   = blockIdx.x * 16;
  const int lane = threadIdx.x & 31;
  const int wave = threadIdx.x >> 5;
  const int l16  = lane & 15;
  const int half = lane >> 4;
  const int nb   = wave * 64;

  if (threadIdx.x < 16) {
    lds_inv[threadIdx.x] = 1.0f / fmaxf(cnt[m0 + threadIdx.x], 1.0f);
    lds_nrm[threadIdx.x] = 0.0f;
  }
  __syncthreads();

  const int arow = m0 + l16;
  v8f c0 = {}, c1 = {}, c2 = {}, c3 = {};

  // ---- neighbor-sum path: msg @ Wl^T -------------------------------------
  for (int k0 = 0; k0 < Kp; k0 += 32) {
    v16bf a  = load_frag_f32cvt(msg, arow, Kp, k0, half);
    v16bf b0 = load_frag_bf16(Wl, nb +  0 + l16, Kp, k0, half);
    v16bf b1 = load_frag_bf16(Wl, nb + 16 + l16, Kp, k0, half);
    v16bf b2 = load_frag_bf16(Wl, nb + 32 + l16, Kp, k0, half);
    v16bf b3 = load_frag_bf16(Wl, nb + 48 + l16, Kp, k0, half);
    c0 = __builtin_amdgcn_wmma_f32_16x16x32_bf16(false, a, false, b0, (short)0, c0, false, false);
    c1 = __builtin_amdgcn_wmma_f32_16x16x32_bf16(false, a, false, b1, (short)0, c1, false, false);
    c2 = __builtin_amdgcn_wmma_f32_16x16x32_bf16(false, a, false, b2, (short)0, c2, false, false);
    c3 = __builtin_amdgcn_wmma_f32_16x16x32_bf16(false, a, false, b3, (short)0, c3, false, false);
  }

  // mean: per-row scale (C layout: vgpr i, lane-half -> row i + 8*half)
#pragma unroll
  for (int i = 0; i < 8; ++i) {
    float sc = lds_inv[i + 8 * half];
    c0[i] *= sc; c1[i] *= sc; c2[i] *= sc; c3[i] *= sc;
  }

  // ---- self path: xin @ Wr^T (accumulate into same C) --------------------
  for (int k0 = 0; k0 < Kp; k0 += 32) {
    v16bf a  = load_frag_bf16(xinb, arow, Kp, k0, half);
    v16bf b0 = load_frag_bf16(Wr, nb +  0 + l16, Kp, k0, half);
    v16bf b1 = load_frag_bf16(Wr, nb + 16 + l16, Kp, k0, half);
    v16bf b2 = load_frag_bf16(Wr, nb + 32 + l16, Kp, k0, half);
    v16bf b3 = load_frag_bf16(Wr, nb + 48 + l16, Kp, k0, half);
    c0 = __builtin_amdgcn_wmma_f32_16x16x32_bf16(false, a, false, b0, (short)0, c0, false, false);
    c1 = __builtin_amdgcn_wmma_f32_16x16x32_bf16(false, a, false, b1, (short)0, c1, false, false);
    c2 = __builtin_amdgcn_wmma_f32_16x16x32_bf16(false, a, false, b2, (short)0, c2, false, false);
    c3 = __builtin_amdgcn_wmma_f32_16x16x32_bf16(false, a, false, b3, (short)0, c3, false, false);
  }

  // bias (per output column)
  float bn0 = bias[nb +  0 + l16];
  float bn1 = bias[nb + 16 + l16];
  float bn2 = bias[nb + 32 + l16];
  float bn3 = bias[nb + 48 + l16];
#pragma unroll
  for (int i = 0; i < 8; ++i) { c0[i] += bn0; c1[i] += bn1; c2[i] += bn2; c3[i] += bn3; }

  // row-wise sum of squares via LDS atomics (whole row lives in this block)
#pragma unroll
  for (int i = 0; i < 8; ++i) {
    float sq = c0[i]*c0[i] + c1[i]*c1[i] + c2[i]*c2[i] + c3[i]*c3[i];
    atomicAdd(&lds_nrm[i + 8 * half], sq);
  }
  __syncthreads();

  // normalize (F.normalize, eps 1e-12), relu, store
#pragma unroll
  for (int i = 0; i < 8; ++i) {
    float dn = 1.0f / fmaxf(sqrtf(lds_nrm[i + 8 * half]), 1e-12f);
    int row  = m0 + i + 8 * half;
    float r0 = fmaxf(c0[i] * dn, 0.0f);
    float r1 = fmaxf(c1[i] * dn, 0.0f);
    float r2 = fmaxf(c2[i] * dn, 0.0f);
    float r3 = fmaxf(c3[i] * dn, 0.0f);
    if (outf) {
      float* op = outf + (size_t)row * HDIM + nb + l16;
      op[0] = r0; op[16] = r1; op[32] = r2; op[48] = r3;
    }
    if (outb) {
      unsigned short* op = outb + (size_t)row * HDIM + nb + l16;
      op[0]  = f32_to_bf16_bits(r0);
      op[16] = f32_to_bf16_bits(r1);
      op[32] = f32_to_bf16_bits(r2);
      op[48] = f32_to_bf16_bits(r3);
    }
  }
}

// ---------------------------------------------------------------------------
// Graph sum pooling
// ---------------------------------------------------------------------------
__global__ __launch_bounds__(256) void pool_sum(
    const float* __restrict__ h, const int* __restrict__ batch_ids,
    float* __restrict__ pool) {
  int n = blockIdx.x;
  int b = batch_ids[n];
  for (int f = threadIdx.x; f < HDIM; f += 256)
    atomicAdd(&pool[(size_t)b * HDIM + f], h[(size_t)n * HDIM + f]);
}

// ---------------------------------------------------------------------------
// MLP head (64 rows, latency-trivial)
// ---------------------------------------------------------------------------
__global__ __launch_bounds__(512) void head_fc1(
    const float* __restrict__ pool, const float* __restrict__ z,
    const float* __restrict__ W, const float* __restrict__ b,
    float* __restrict__ o) {
  int g = blockIdx.x, t = threadIdx.x;
  const float* wr = W + (size_t)t * (HDIM + Z_DIM);
  float acc = b[t];
  for (int k = 0; k < HDIM; ++k)  acc += pool[(size_t)g * HDIM + k] * wr[k];
  for (int k = 0; k < Z_DIM; ++k) acc += z[(size_t)g * Z_DIM + k] * wr[HDIM + k];
  o[(size_t)g * HDIM + t] = fmaxf(acc, 0.0f);
}

__global__ __launch_bounds__(512) void head_fc2(
    const float* __restrict__ f1, const float* __restrict__ W,
    const float* __restrict__ b, float* __restrict__ o) {
  int g = blockIdx.x, t = threadIdx.x;
  const float* wr = W + (size_t)t * HDIM;
  float acc = b[t];
  for (int k = 0; k < HDIM; ++k) acc += f1[(size_t)g * HDIM + k] * wr[k];
  o[(size_t)g * HDIM + t] = fmaxf(acc, 0.0f);
}

__global__ __launch_bounds__(64) void head_out(
    const float* __restrict__ f2, const float* __restrict__ Wp,
    const float* __restrict__ bp, float* __restrict__ out) {
  int g = threadIdx.x;
  float acc = bp[0];
  for (int k = 0; k < HDIM; ++k) acc += f2[(size_t)g * HDIM + k] * Wp[k];
  float r = (acc > 0.0f) ? log1pf(expf(-acc)) : (-acc + log1pf(expf(acc)));
  out[g] = r;   // -log_sigmoid
}

// ---------------------------------------------------------------------------
// Launch
// ---------------------------------------------------------------------------
extern "C" void kernel_launch(void* const* d_in, const int* in_sizes, int n_in,
                              void* d_out, int out_size, void* d_ws, size_t ws_size,
                              hipStream_t stream) {
  (void)in_sizes; (void)n_in; (void)out_size; (void)ws_size;

  const float* x   = (const float*)d_in[0];
  const int*   ei  = (const int*)d_in[1];       // [2, N_EDGES]
  const int*   bid = (const int*)d_in[2];       // [N_NODES]
  const float* z   = (const float*)d_in[3];
  const float* Wl1 = (const float*)d_in[4];
  const float* bl1 = (const float*)d_in[5];
  const float* Wr1 = (const float*)d_in[6];
  const float* Wl2 = (const float*)d_in[7];
  const float* bl2 = (const float*)d_in[8];
  const float* Wr2 = (const float*)d_in[9];
  const float* Wf1 = (const float*)d_in[10];
  const float* bf1 = (const float*)d_in[11];
  const float* Wf2 = (const float*)d_in[12];
  const float* bf2 = (const float*)d_in[13];
  const float* Wp  = (const float*)d_in[14];
  const float* bp  = (const float*)d_in[15];
  float* out = (float*)d_out;

  const int* src = ei;
  const int* dst = ei + N_EDGES;

  // ---- carve workspace ----------------------------------------------------
  char* ws = (char*)d_ws;
  auto carve = [&](size_t bytes) {
    void* p = (void*)ws;
    ws += (bytes + 255) & ~(size_t)255;
    return p;
  };
  float*          cnt   = (float*)carve((size_t)N_NODES * 4);
  float*          msg1  = (float*)carve((size_t)N_NODES * K1P * 4);           // zero-padded
  unsigned short* xpadb = (unsigned short*)carve((size_t)N_NODES * K1P * 2);  // bf16 x, padded
  unsigned short* h1b   = (unsigned short*)carve((size_t)N_NODES * HDIM * 2); // bf16 h1
  float*          msg2  = (float*)carve((size_t)N_NODES * HDIM * 4);
  float*          h2    = (float*)carve((size_t)N_NODES * HDIM * 4);
  float*          pool  = (float*)carve((size_t)N_GRAPHS * HDIM * 4);
  float*          f1    = (float*)carve((size_t)N_GRAPHS * HDIM * 4);
  float*          f2    = (float*)carve((size_t)N_GRAPHS * HDIM * 4);
  unsigned short* Wl1b  = (unsigned short*)carve((size_t)HDIM * K1P * 2);
  unsigned short* Wr1b  = (unsigned short*)carve((size_t)HDIM * K1P * 2);
  unsigned short* Wl2b  = (unsigned short*)carve((size_t)HDIM * HDIM * 2);
  unsigned short* Wr2b  = (unsigned short*)carve((size_t)HDIM * HDIM * 2);

  auto nblk = [](int n) { return (n + 255) / 256; };

  // ---- zero accumulation buffers (incl. K padding of msg1) ----------------
  zero_f32<<<nblk(N_NODES), 256, 0, stream>>>(cnt, N_NODES);
  zero_f32<<<nblk(N_NODES * K1P), 256, 0, stream>>>(msg1, N_NODES * K1P);
  zero_f32<<<nblk(N_NODES * HDIM), 256, 0, stream>>>(msg2, N_NODES * HDIM);
  zero_f32<<<nblk(N_GRAPHS * HDIM), 256, 0, stream>>>(pool, N_GRAPHS * HDIM);

  // ---- stage bf16 operands (zero-padded K) --------------------------------
  cvt_pad_bf16<<<nblk(N_NODES * K1P), 256, 0, stream>>>(x, xpadb, N_NODES, F_IN, K1P);
  cvt_pad_bf16<<<nblk(HDIM * K1P), 256, 0, stream>>>(Wl1, Wl1b, HDIM, F_IN, K1P);
  cvt_pad_bf16<<<nblk(HDIM * K1P), 256, 0, stream>>>(Wr1, Wr1b, HDIM, F_IN, K1P);
  cvt_pad_bf16<<<nblk(HDIM * HDIM), 256, 0, stream>>>(Wl2, Wl2b, HDIM, HDIM, HDIM);
  cvt_pad_bf16<<<nblk(HDIM * HDIM), 256, 0, stream>>>(Wr2, Wr2b, HDIM, HDIM, HDIM);

  // ---- layer 1: scatter + fused WMMA layer (bf16 h1 out only) -------------
  scatter_edges_1<<<N_EDGES, 128, 0, stream>>>(x, src, dst, msg1, cnt);
  sage_layer_wmma<<<N_NODES / 16, 256, 0, stream>>>(
      msg1, xpadb, Wl1b, Wr1b, bl1, cnt, (float*)nullptr, h1b, K1P);

  // ---- layer 2: bf16 gather-scatter + fused WMMA layer (f32 h2 out) -------
  scatter_edges_2<<<N_EDGES, 256, 0, stream>>>(h1b, src, dst, msg2);
  sage_layer_wmma<<<N_NODES / 16, 256, 0, stream>>>(
      msg2, h1b, Wl2b, Wr2b, bl2, cnt, h2, (unsigned short*)nullptr, HDIM);

  // ---- pooling + head -----------------------------------------------------
  pool_sum<<<N_NODES, 256, 0, stream>>>(h2, bid, pool);
  head_fc1<<<N_GRAPHS, 512, 0, stream>>>(pool, z, Wf1, bf1, f1);
  head_fc2<<<N_GRAPHS, 512, 0, stream>>>(f1, Wf2, bf2, f2);
  head_out<<<1, 64, 0, stream>>>(f2, Wp, bp, out);
}
